// AFIA_73151882986116
// MI455X (gfx1250) — matrix operand order repaired and checked by reading, not compiled, not run on hardware
//
#include <hip/hip_runtime.h>
#include <hip/hip_bf16.h>

typedef __attribute__((ext_vector_type(16))) _Float16 v16h;
typedef __attribute__((ext_vector_type(8)))  _Float16 v8h;
typedef __attribute__((ext_vector_type(8)))  float    v8f;
typedef __attribute__((ext_vector_type(4)))  int      v4i;

#define DEVINL __device__ __forceinline__

// gfx1250 async global->LDS path (both builtins confirmed present; signature is
// (int4 addrspace(1)*, int4 addrspace(3)*, imm offset, imm cpol)).
#if defined(__has_builtin)
#if __has_builtin(__builtin_amdgcn_global_load_async_to_lds_b128) && \
    __has_builtin(__builtin_amdgcn_s_wait_asynccnt)
#define HAVE_ASYNC_LDS 1
#endif
#endif

#if defined(HAVE_ASYNC_LDS)
typedef __attribute__((address_space(1))) v4i as1_v4i;
typedef __attribute__((address_space(3))) v4i as3_v4i;
DEVINL void async_b128(const _Float16* g, _Float16* l) {
  __builtin_amdgcn_global_load_async_to_lds_b128(
      (as1_v4i*)(void*)g, (as3_v4i*)(void*)l, 0, 0);
}
#endif

// ---------------------------------------------------------------------------
// WMMA helper: D = A(16x32 f16) * B(32x16 f16) + C(16x16 f32)
// ---------------------------------------------------------------------------
DEVINL v8f wmma_f16(v16h a, v16h b, v8f c) {
  return __builtin_amdgcn_wmma_f32_16x16x32_f16(
      /*neg_a=*/false, a, /*neg_b=*/false, b,
      /*c_mod=*/(short)0, c, /*reuse_a=*/false, /*reuse_b=*/false);
}

// A-fragment (16x32) from row-major f16: lanes 0-15 row M=lane, halves 0..15 ->
// K 0..15; lanes 16-31 K offset +16.  Two 16B loads.
DEVINL v16h fragA_rowmajor(const _Float16* base, int stride) {
  const int lane = threadIdx.x & 31;
  const _Float16* p = base + (size_t)(lane & 15) * stride + ((lane >> 4) << 4);
  v8h lo = *reinterpret_cast<const v8h*>(p);
  v8h hi = *reinterpret_cast<const v8h*>(p + 8);
  v16h r;
#pragma unroll
  for (int i = 0; i < 8; ++i) { r[i] = lo[i]; r[i + 8] = hi[i]; }
  return r;
}

// B-fragment (32x16) with element (k,n) at rowbase + n*stride + k.
DEVINL v16h fragB_contig(const _Float16* rowbase, int stride) {
  const int lane = threadIdx.x & 31;
  const _Float16* q = rowbase + (size_t)(lane & 15) * stride + ((lane >> 4) << 4);
  v8h lo = *reinterpret_cast<const v8h*>(q);
  v8h hi = *reinterpret_cast<const v8h*>(q + 8);
  v16h r;
#pragma unroll
  for (int i = 0; i < 8; ++i) { r[i] = lo[i]; r[i + 8] = hi[i]; }
  return r;
}

// B-fragment from pre-packed weights: each lane owns 16 contiguous halves.
DEVINL v16h fragB_packed(const _Float16* p) {
  const int lane = threadIdx.x & 31;
  const _Float16* q = p + lane * 16;
  v8h lo = *reinterpret_cast<const v8h*>(q);
  v8h hi = *reinterpret_cast<const v8h*>(q + 8);
  v16h r;
#pragma unroll
  for (int i = 0; i < 8; ++i) { r[i] = lo[i]; r[i + 8] = hi[i]; }
  return r;
}

// Row reductions across 16-lane half-wave groups (wave32).
DEVINL float red_max16(float v) {
#pragma unroll
  for (int m = 1; m < 16; m <<= 1) v = fmaxf(v, __shfl_xor(v, m, 32));
  return v;
}
DEVINL float red_sum16(float v) {
#pragma unroll
  for (int m = 1; m < 16; m <<= 1) v += __shfl_xor(v, m, 32);
  return v;
}

// ---------------------------------------------------------------------------
// Weight pre-pack into WMMA B-fragment layout.
// mode 0: 1x1 (O,I); mode 1: conv3x3 (O,I,3,3); mode 2: deconv3x3 (I,O,3,3).
// ---------------------------------------------------------------------------
__global__ void prep_w_kernel(const float* __restrict__ src,
                              _Float16* __restrict__ dst, int mode) {
  const int taps = (mode == 0) ? 1 : 9;
  const int total = taps * 4 * 8 * 32 * 16;
  for (int e = blockIdx.x * blockDim.x + threadIdx.x; e < total;
       e += gridDim.x * blockDim.x) {
    int i    = e & 15;
    int lane = (e >> 4) & 31;
    int ot   = (e >> 9) & 7;
    int ks   = (e >> 12) & 3;
    int tap  = e >> 14;
    int o = ot * 16 + (lane & 15);
    int c = ks * 32 + ((lane >> 4) << 4) + i;
    float v;
    if (mode == 0) {
      v = src[o * 128 + c];
    } else if (mode == 1) {
      int kh = tap / 3, kw = tap % 3;
      v = src[((o * 128 + c) * 3 + kh) * 3 + kw];
    } else {
      int kh = tap / 3, kw = tap % 3;
      v = src[((c * 128 + o) * 3 + (2 - kh)) * 3 + (2 - kw)];
    }
    dst[e] = (_Float16)v;
  }
}

// ---------------------------------------------------------------------------
// LayerNorm1 + NCHW -> token-major f16 + channel shuffle (groups=4).
// ---------------------------------------------------------------------------
__global__ void ln1_kernel(const float* __restrict__ x,
                           const float* __restrict__ gam,
                           const float* __restrict__ bet,
                           _Float16* __restrict__ xn_h,
                           _Float16* __restrict__ xs_h) {
  const int tok = blockIdx.x;
  const int c = threadIdx.x;
  const int bi = tok >> 12;
  const int sp = tok & 4095;
  float v = x[((size_t)bi * 128 + c) * 4096 + sp];
  __shared__ float s1[128], s2[128], sv[128];
  s1[c] = v; s2[c] = v * v;
  __syncthreads();
  for (int off = 64; off > 0; off >>= 1) {
    if (c < off) { s1[c] += s1[c + off]; s2[c] += s2[c + off]; }
    __syncthreads();
  }
  float mean = s1[0] * (1.0f / 128.0f);
  float var  = s2[0] * (1.0f / 128.0f) - mean * mean;
  float rs = rsqrtf(var + 1e-5f);
  float xn = (v - mean) * rs * gam[c] + bet[c];
  sv[c] = xn;
  __syncthreads();
  xn_h[(size_t)tok * 128 + c] = (_Float16)xn;
  xs_h[(size_t)tok * 128 + c] = (_Float16)sv[(c & 3) * 32 + (c >> 2)];
}

// ---------------------------------------------------------------------------
// 1x1 conv as GEMM.  One wave per 32-token tile: each B fragment feeds two
// independent WMMAs (A reuse across the N loop, B reuse across the M tiles).
// ---------------------------------------------------------------------------
__global__ void __launch_bounds__(32)
gemm1x1_kernel(const _Float16* __restrict__ in, const _Float16* __restrict__ wpk,
               const float* __restrict__ bias, _Float16* __restrict__ out_h,
               float* __restrict__ out_f, int out_is_f32) {
  const int tok0 = blockIdx.x * 32;
  const int lane = threadIdx.x & 31;
  const int col = lane & 15;
  const int rbase = (lane >> 4) << 3;

  v8f acc0[8], acc1[8];
#pragma unroll
  for (int t = 0; t < 8; ++t) {
    float bb = bias[t * 16 + col];
#pragma unroll
    for (int r = 0; r < 8; ++r) { acc0[t][r] = bb; acc1[t][r] = bb; }
  }
#pragma unroll
  for (int ks = 0; ks < 4; ++ks) {
    v16h a0 = fragA_rowmajor(in + (size_t)tok0 * 128 + ks * 32, 128);
    v16h a1 = fragA_rowmajor(in + (size_t)(tok0 + 16) * 128 + ks * 32, 128);
#pragma unroll
    for (int t = 0; t < 8; ++t) {
      v16h b = fragB_packed(wpk + (size_t)(ks * 8 + t) * 512);
      acc0[t] = wmma_f16(a0, b, acc0[t]);
      acc1[t] = wmma_f16(a1, b, acc1[t]);
    }
  }
#pragma unroll
  for (int t = 0; t < 8; ++t)
#pragma unroll
    for (int r = 0; r < 8; ++r) {
      size_t i0 = (size_t)(tok0 + rbase + r) * 128 + t * 16 + col;
      size_t i1 = (size_t)(tok0 + 16 + rbase + r) * 128 + t * 16 + col;
      if (out_is_f32) { out_f[i0] = acc0[t][r]; out_f[i1] = acc1[t][r]; }
      else { out_h[i0] = (_Float16)acc0[t][r]; out_h[i1] = (_Float16)acc1[t][r]; }
    }
}

// ---------------------------------------------------------------------------
// 3x3 conv (pad 1) as 9-tap accumulated GEMM on 32-token tiles (one image row,
// w0 in {0,32}); per-lane masking in w, per-tap masking in h.
// ---------------------------------------------------------------------------
DEVINL v16h fragA_shift(const _Float16* in, int tok0, int w0, int rofs,
                        int dh, int dw, int c0) {
  const int lane = threadIdx.x & 31;
  const int r = lane & 15;
  const int w = w0 + rofs + r + dw;
  v16h out;
  if (w < 0 || w >= 64) {
#pragma unroll
    for (int i = 0; i < 16; ++i) out[i] = (_Float16)0.0f;
    return out;
  }
  const _Float16* p =
      in + (size_t)(tok0 + rofs + r + dh * 64 + dw) * 128 + c0 + ((lane >> 4) << 4);
  v8h lo = *reinterpret_cast<const v8h*>(p);
  v8h hi = *reinterpret_cast<const v8h*>(p + 8);
#pragma unroll
  for (int i = 0; i < 8; ++i) { out[i] = lo[i]; out[i + 8] = hi[i]; }
  return out;
}

__global__ void __launch_bounds__(32)
conv3x3_kernel(const _Float16* __restrict__ in, const _Float16* __restrict__ wpk,
               const float* __restrict__ bias, _Float16* __restrict__ out_h,
               float* __restrict__ out_f, int do_relu, int out_is_f32) {
  const int tok0 = blockIdx.x * 32;
  const int lane = threadIdx.x & 31;
  const int col = lane & 15;
  const int rbase = (lane >> 4) << 3;
  const int hh = (tok0 >> 6) & 63;
  const int w0 = tok0 & 63;

  v8f acc0[8], acc1[8];
#pragma unroll
  for (int t = 0; t < 8; ++t) {
    float bb = bias[t * 16 + col];
#pragma unroll
    for (int r = 0; r < 8; ++r) { acc0[t][r] = bb; acc1[t][r] = bb; }
  }

  for (int tap = 0; tap < 9; ++tap) {
    const int dh = tap / 3 - 1;
    const int dw = tap % 3 - 1;
    if (hh + dh < 0 || hh + dh >= 64) continue;
#pragma unroll
    for (int ks = 0; ks < 4; ++ks) {
      v16h a0 = fragA_shift(in, tok0, w0, 0,  dh, dw, ks * 32);
      v16h a1 = fragA_shift(in, tok0, w0, 16, dh, dw, ks * 32);
      const _Float16* wp = wpk + (size_t)((tap * 4 + ks) * 8) * 512;
#pragma unroll
      for (int t = 0; t < 8; ++t) {
        v16h b = fragB_packed(wp + (size_t)t * 512);
        acc0[t] = wmma_f16(a0, b, acc0[t]);
        acc1[t] = wmma_f16(a1, b, acc1[t]);
      }
    }
  }

#pragma unroll
  for (int t = 0; t < 8; ++t)
#pragma unroll
    for (int r = 0; r < 8; ++r) {
      float v0 = acc0[t][r], v1 = acc1[t][r];
      if (do_relu) { v0 = fmaxf(v0, 0.0f); v1 = fmaxf(v1, 0.0f); }
      size_t i0 = (size_t)(tok0 + rbase + r) * 128 + t * 16 + col;
      size_t i1 = (size_t)(tok0 + 16 + rbase + r) * 128 + t * 16 + col;
      if (out_is_f32) { out_f[i0] = v0; out_f[i1] = v1; }
      else { out_h[i0] = (_Float16)v0; out_h[i1] = (_Float16)v1; }
    }
}

// ---------------------------------------------------------------------------
// Flash attention: 1 wave per 16-query tile.  K/V chunks of 32 keys staged in
// double-buffered LDS via async global->LDS (fallback: vector copies), so the
// next chunk's fill overlaps the current chunk's 16 WMMAs + softmax.
// ---------------------------------------------------------------------------
DEVINL void stage_kv(const _Float16* kp, const _Float16* vp,
                     _Float16* dk, _Float16* dv) {
  const int lane = threadIdx.x & 31;
#if defined(HAVE_ASYNC_LDS)
#pragma unroll
  for (int j = 0; j < 8; ++j) {
    async_b128(kp + j * 8, dk + lane * 128 + j * 8);
    async_b128(vp + j * 8, dv + lane * 128 + j * 8);
  }
#else
#pragma unroll
  for (int j = 0; j < 8; ++j) {
    *reinterpret_cast<v8h*>(dk + lane * 128 + j * 8) =
        *reinterpret_cast<const v8h*>(kp + j * 8);
    *reinterpret_cast<v8h*>(dv + lane * 128 + j * 8) =
        *reinterpret_cast<const v8h*>(vp + j * 8);
  }
#endif
}

DEVINL void wait_stage() {
#if defined(HAVE_ASYNC_LDS)
  __builtin_amdgcn_s_wait_asynccnt(0);
#endif
  __syncthreads();
}

__global__ void __launch_bounds__(32)
attn_kernel(const _Float16* __restrict__ Q, const _Float16* __restrict__ K,
            const _Float16* __restrict__ V, _Float16* __restrict__ Oh) {
  const int tile = blockIdx.x;     // 0..1023
  const int bi = tile >> 8;
  const int q0 = bi * 4096 + (tile & 255) * 16;
  const int kbase = bi * 4096;
  const int lane = threadIdx.x & 31;
  const int col = lane & 15;
  const int rbase = (lane >> 4) << 3;
  const int kb16 = (lane >> 4) << 4;

  __shared__ _Float16 lk[2][32 * 128];
  __shared__ _Float16 lv[2][32 * 128];
  __shared__ _Float16 lp[16 * 32];

  v16h qf[4];
#pragma unroll
  for (int ks = 0; ks < 4; ++ks)
    qf[ks] = fragA_rowmajor(Q + (size_t)q0 * 128 + ks * 32, 128);

  v8f o[8];
#pragma unroll
  for (int t = 0; t < 8; ++t)
#pragma unroll
    for (int r = 0; r < 8; ++r) o[t][r] = 0.0f;
  float m[8], l[8];
#pragma unroll
  for (int r = 0; r < 8; ++r) { m[r] = -1e30f; l[r] = 0.0f; }

  const float scale = 0.08838834764831845f;  // 1/sqrt(128)

  // prologue: stage chunk 0
  {
    const _Float16* kp = K + (size_t)(kbase + lane) * 128;
    const _Float16* vp = V + (size_t)(kbase + lane) * 128;
    stage_kv(kp, vp, lk[0], lv[0]);
    wait_stage();
  }

  for (int kc = 0; kc < 128; ++kc) {
    const int cur = kc & 1;
    const int nxt = cur ^ 1;
    // overlap: fill next chunk while computing on current one.
    if (kc + 1 < 128) {
      const _Float16* kp = K + (size_t)(kbase + (kc + 1) * 32 + lane) * 128;
      const _Float16* vp = V + (size_t)(kbase + (kc + 1) * 32 + lane) * 128;
      __builtin_prefetch(kp + 32 * 128, 0, 1);   // warm L2 two chunks ahead
      __builtin_prefetch(vp + 32 * 128, 0, 1);
      stage_kv(kp, vp, lk[nxt], lv[nxt]);
    }
    const _Float16* kcur = lk[cur];
    const _Float16* vcur = lv[cur];

    // S = Q * K^T for 32 keys -> two 16x16 f32 fragments.
    v8f s[2];
#pragma unroll
    for (int f = 0; f < 2; ++f) {
#pragma unroll
      for (int r = 0; r < 8; ++r) s[f][r] = 0.0f;
#pragma unroll
      for (int ks = 0; ks < 4; ++ks)
        s[f] = wmma_f16(qf[ks],
                        fragB_contig(kcur + (size_t)f * 16 * 128 + ks * 32, 128),
                        s[f]);
#pragma unroll
      for (int r = 0; r < 8; ++r) s[f][r] *= scale;
    }

    // online softmax row stats (rows live in 16-lane groups).
    float mnew[8], alpha[8];
#pragma unroll
    for (int r = 0; r < 8; ++r) {
      float t = red_max16(fmaxf(s[0][r], s[1][r]));
      mnew[r] = fmaxf(m[r], t);
      alpha[r] = __expf(m[r] - mnew[r]);
    }
#pragma unroll
    for (int f = 0; f < 2; ++f)
#pragma unroll
      for (int r = 0; r < 8; ++r) s[f][r] = __expf(s[f][r] - mnew[r]);
#pragma unroll
    for (int r = 0; r < 8; ++r) {
      float t = red_sum16(s[0][r] + s[1][r]);
      l[r] = l[r] * alpha[r] + t;
      m[r] = mnew[r];
#pragma unroll
      for (int t8 = 0; t8 < 8; ++t8) o[t8][r] *= alpha[r];
    }

    // transpose P (C-layout) -> A-layout through LDS.
#pragma unroll
    for (int f = 0; f < 2; ++f)
#pragma unroll
      for (int r = 0; r < 8; ++r)
        lp[(r + rbase) * 32 + col + f * 16] = (_Float16)s[f][r];
    __syncthreads();
    v16h pa = fragA_rowmajor(lp, 32);

    // O += P * V  (8 channel tiles of 16).
#pragma unroll
    for (int t8 = 0; t8 < 8; ++t8) {
      v16h bv;
#pragma unroll
      for (int i = 0; i < 16; ++i)
        bv[i] = vcur[(size_t)(kb16 + i) * 128 + t8 * 16 + col];
      o[t8] = wmma_f16(pa, bv, o[t8]);
    }
    wait_stage();   // async chunk fill + LDS ordering for next iteration
  }

  // epilogue: normalize, ReLU, store f16 token-major.
#pragma unroll
  for (int t8 = 0; t8 < 8; ++t8)
#pragma unroll
    for (int r = 0; r < 8; ++r) {
      float vv = fmaxf(o[t8][r] / l[r], 0.0f);
      size_t idx = (size_t)(q0 + rbase + r) * 128 + t8 * 16 + col;
      Oh[idx] = (_Float16)vv;
    }
}

// ---------------------------------------------------------------------------
// Final: out = LN2(wo_out + sc + residual), back to NCHW f32.
// ---------------------------------------------------------------------------
__global__ void final_ln2_kernel(const float* __restrict__ a,
                                 const float* __restrict__ sc,
                                 const float* __restrict__ x,
                                 const float* __restrict__ gam,
                                 const float* __restrict__ bet,
                                 float* __restrict__ out) {
  const int tok = blockIdx.x;
  const int c = threadIdx.x;
  const int bi = tok >> 12;
  const int sp = tok & 4095;
  float v = a[(size_t)tok * 128 + c] + sc[(size_t)tok * 128 + c] +
            x[((size_t)bi * 128 + c) * 4096 + sp];
  __shared__ float s1[128], s2[128];
  s1[c] = v; s2[c] = v * v;
  __syncthreads();
  for (int off = 64; off > 0; off >>= 1) {
    if (c < off) { s1[c] += s1[c + off]; s2[c] += s2[c + off]; }
    __syncthreads();
  }
  float mean = s1[0] * (1.0f / 128.0f);
  float var  = s2[0] * (1.0f / 128.0f) - mean * mean;
  float rs = rsqrtf(var + 1e-5f);
  out[((size_t)bi * 128 + c) * 4096 + sp] = (v - mean) * rs * gam[c] + bet[c];
}

// ---------------------------------------------------------------------------
extern "C" void kernel_launch(void* const* d_in, const int* in_sizes, int n_in,
                              void* d_out, int out_size, void* d_ws, size_t ws_size,
                              hipStream_t stream) {
  const float* x     = (const float*)d_in[0];
  const float* ln1_g = (const float*)d_in[1];
  const float* ln1_b = (const float*)d_in[2];
  const float* wq1   = (const float*)d_in[3];
  const float* bq1   = (const float*)d_in[4];
  const float* wqd   = (const float*)d_in[5];
  const float* bqd   = (const float*)d_in[6];
  const float* wk1   = (const float*)d_in[7];
  const float* bk1   = (const float*)d_in[8];
  const float* wkd   = (const float*)d_in[9];
  const float* bkd   = (const float*)d_in[10];
  const float* wv1   = (const float*)d_in[11];
  const float* bv1   = (const float*)d_in[12];
  const float* wvd   = (const float*)d_in[13];
  const float* bvd   = (const float*)d_in[14];
  const float* ws1   = (const float*)d_in[15];
  const float* bs1   = (const float*)d_in[16];
  const float* ws2   = (const float*)d_in[17];
  const float* bs2   = (const float*)d_in[18];
  const float* wo    = (const float*)d_in[19];
  const float* bo    = (const float*)d_in[20];
  const float* ln2_g = (const float*)d_in[21];
  const float* ln2_b = (const float*)d_in[22];
  float* out = (float*)d_out;

  char* ws = (char*)d_ws;
  const size_t NT = 16384;                        // tokens
  const size_t FH = NT * 128 * sizeof(_Float16);  // 4 MiB per f16 tensor
  const size_t W1 = 16384 * sizeof(_Float16);
  const size_t W3 = 147456 * sizeof(_Float16);
  const size_t needed = 14 * FH + 4 * W1 + 5 * W3;
  if (ws_size < needed) return;

  _Float16* xn_h = (_Float16*)(ws + 0 * FH);
  _Float16* xs_h = (_Float16*)(ws + 1 * FH);
  _Float16* q1   = (_Float16*)(ws + 2 * FH);
  _Float16* k1   = (_Float16*)(ws + 3 * FH);
  _Float16* v1   = (_Float16*)(ws + 4 * FH);
  _Float16* Qd   = (_Float16*)(ws + 5 * FH);
  _Float16* Kd   = (_Float16*)(ws + 6 * FH);
  _Float16* Vd   = (_Float16*)(ws + 7 * FH);
  _Float16* at_h = (_Float16*)(ws + 8 * FH);
  _Float16* s1_h = (_Float16*)(ws + 9 * FH);
  float*    wo_o = (float*)   (ws + 10 * FH);
  float*    sc_o = (float*)   (ws + 12 * FH);
  char* wb = ws + 14 * FH;
  _Float16* pw_q1 = (_Float16*)(wb + 0 * W1);
  _Float16* pw_k1 = (_Float16*)(wb + 1 * W1);
  _Float16* pw_v1 = (_Float16*)(wb + 2 * W1);
  _Float16* pw_o  = (_Float16*)(wb + 3 * W1);
  _Float16* pw_qd = (_Float16*)(wb + 4 * W1 + 0 * W3);
  _Float16* pw_kd = (_Float16*)(wb + 4 * W1 + 1 * W3);
  _Float16* pw_vd = (_Float16*)(wb + 4 * W1 + 2 * W3);
  _Float16* pw_s1 = (_Float16*)(wb + 4 * W1 + 3 * W3);
  _Float16* pw_s2 = (_Float16*)(wb + 4 * W1 + 4 * W3);

  prep_w_kernel<<<64, 256, 0, stream>>>(wq1, pw_q1, 0);
  prep_w_kernel<<<64, 256, 0, stream>>>(wk1, pw_k1, 0);
  prep_w_kernel<<<64, 256, 0, stream>>>(wv1, pw_v1, 0);
  prep_w_kernel<<<64, 256, 0, stream>>>(wo,  pw_o,  0);
  prep_w_kernel<<<256, 256, 0, stream>>>(wqd, pw_qd, 2);
  prep_w_kernel<<<256, 256, 0, stream>>>(wkd, pw_kd, 2);
  prep_w_kernel<<<256, 256, 0, stream>>>(wvd, pw_vd, 2);
  prep_w_kernel<<<256, 256, 0, stream>>>(ws1, pw_s1, 1);
  prep_w_kernel<<<256, 256, 0, stream>>>(ws2, pw_s2, 1);

  ln1_kernel<<<16384, 128, 0, stream>>>(x, ln1_g, ln1_b, xn_h, xs_h);

  gemm1x1_kernel<<<512, 32, 0, stream>>>(xn_h, pw_q1, bq1, q1, nullptr, 0);
  gemm1x1_kernel<<<512, 32, 0, stream>>>(xn_h, pw_k1, bk1, k1, nullptr, 0);
  gemm1x1_kernel<<<512, 32, 0, stream>>>(xn_h, pw_v1, bv1, v1, nullptr, 0);

  conv3x3_kernel<<<512, 32, 0, stream>>>(q1, pw_qd, bqd, Qd, nullptr, 0, 0);
  conv3x3_kernel<<<512, 32, 0, stream>>>(k1, pw_kd, bkd, Kd, nullptr, 0, 0);
  conv3x3_kernel<<<512, 32, 0, stream>>>(v1, pw_vd, bvd, Vd, nullptr, 0, 0);

  attn_kernel<<<1024, 32, 0, stream>>>(Qd, Kd, Vd, at_h);

  gemm1x1_kernel<<<512, 32, 0, stream>>>(at_h, pw_o, bo, nullptr, wo_o, 1);

  conv3x3_kernel<<<512, 32, 0, stream>>>(xs_h, pw_s1, bs1, s1_h, nullptr, 1, 0);
  conv3x3_kernel<<<512, 32, 0, stream>>>(s1_h, pw_s2, bs2, nullptr, sc_o, 0, 1);

  final_ln2_kernel<<<16384, 128, 0, stream>>>(wo_o, sc_o, x, ln2_g, ln2_b, out);
}
